// real_fp4linear_28707561406721
// MI455X (gfx1250) — 8-bit matrix operand order repaired and checked by reading, compiled, not run on hardware
//
#include <hip/hip_runtime.h>
#include <hip/hip_bf16.h>

typedef __attribute__((ext_vector_type(16))) int   v16i;
typedef __attribute__((ext_vector_type(4)))  int   v4i;
typedef __attribute__((ext_vector_type(2)))  int   v2i;
typedef __attribute__((ext_vector_type(8)))  float v8f;

#define GEMM_M 32768
#define GEMM_N 1024
#define GEMM_K 1024
#define KG     4
#define LDS_STRIDE 144

__global__ __launch_bounds__(256) void fp4_quant_kernel(
    const float* __restrict__ src, unsigned char* __restrict__ q,
    float* __restrict__ scales, int numGroups)
{
    int gwave = (int)((blockIdx.x * blockDim.x + threadIdx.x) >> 5);
    int lane  = (int)(threadIdx.x & 31u);
    if (gwave >= numGroups) return;

    size_t base = (size_t)gwave * 256 + (size_t)lane * 8;
    float4 x0 = *(const float4*)(src + base);
    float4 x1 = *(const float4*)(src + base + 4);

    float am = fabsf(x0.x);
    am = fmaxf(am, fabsf(x0.y)); am = fmaxf(am, fabsf(x0.z));
    am = fmaxf(am, fabsf(x0.w)); am = fmaxf(am, fabsf(x1.x));
    am = fmaxf(am, fabsf(x1.y)); am = fmaxf(am, fabsf(x1.z));
    am = fmaxf(am, fabsf(x1.w));
    #pragma unroll
    for (int off = 16; off > 0; off >>= 1)
        am = fmaxf(am, __shfl_xor(am, off, 32));

    float scale = fmaxf(am, 1e-6f);
    if (lane == 0) scales[gwave] = scale;
    float inv = 1.0f / scale;

    auto q1 = [&](float x) -> unsigned int {
        float v = fminf(fmaxf(x * inv, -1.0f), 1.0f);
        float a = fabsf(v);
        unsigned int m = (a < 0.25f) ? 0x00u : ((a < 0.75f) ? 0x38u : 0x40u);
        return (v < 0.0f) ? (m | 0x80u) : m;
    };

    unsigned int b0 = q1(x0.x) | (q1(x0.y) << 8) | (q1(x0.z) << 16) | (q1(x0.w) << 24);
    unsigned int b1 = q1(x1.x) | (q1(x1.y) << 8) | (q1(x1.z) << 16) | (q1(x1.w) << 24);
    v2i packed; packed[0] = (int)b0; packed[1] = (int)b1;
    *(v2i*)(q + base) = packed;
}

__global__ __launch_bounds__(256) void fp4_gemm_wmma_kernel(
    const unsigned char* __restrict__ qx, const unsigned char* __restrict__ qw,
    const float* __restrict__ sx, const float* __restrict__ sw,
    const float* __restrict__ bias, float* __restrict__ out)
{
    const int K = GEMM_K, N = GEMM_N;
    const int NCHUNK = GEMM_K / 128;

    __shared__ __align__(16) unsigned char sA[2][128][LDS_STRIDE];
    __shared__ __align__(16) unsigned char sB[2][64][LDS_STRIDE];

    int tid   = (int)threadIdx.x;
    int wave  = tid >> 5;
    int lane  = tid & 31;
    int laneN = lane & 15;
    int half  = lane >> 4;
    int wm    = wave & 3;
    int wn    = wave >> 2;

    int mWG = (int)blockIdx.y * 128;
    int nWG = (int)blockIdx.x * 64;

    unsigned ldsA[2], ldsB[2];
    ldsA[0] = (unsigned)(uintptr_t)&sA[0][0][0];
    ldsA[1] = (unsigned)(uintptr_t)&sA[1][0][0];
    ldsB[0] = (unsigned)(uintptr_t)&sB[0][0][0];
    ldsB[1] = (unsigned)(uintptr_t)&sB[1][0][0];

    auto issue_chunk = [&](int kk, int buf) {
        int k0 = kk * 128;
        #pragma unroll
        for (int r = 0; r < 4; ++r) {
            int c   = tid + r * 256;
            int row = c >> 3;
            int col = (c & 7) * 16;
            unsigned long long gp = (unsigned long long)(uintptr_t)
                (qx + (size_t)(mWG + row) * K + k0 + col);
            unsigned la = ldsA[buf] + (unsigned)(row * LDS_STRIDE + col);
            asm volatile("global_load_async_to_lds_b128 %0, %1, off"
                         :: "v"(la), "v"(gp) : "memory");
        }
        #pragma unroll
        for (int r = 0; r < 2; ++r) {
            int c   = tid + r * 256;
            int row = c >> 3;
            int col = (c & 7) * 16;
            unsigned long long gp = (unsigned long long)(uintptr_t)
                (qw + (size_t)(nWG + row) * K + k0 + col);
            unsigned lb = ldsB[buf] + (unsigned)(row * LDS_STRIDE + col);
            asm volatile("global_load_async_to_lds_b128 %0, %1, off"
                         :: "v"(lb), "v"(gp) : "memory");
        }
    };

    issue_chunk(0, 0);

    v8f facc[2][2] = {};
    v8f acc[2][2];

    for (int kk = 0; kk < NCHUNK; ++kk) {
        int buf = kk & 1;
        asm volatile("s_wait_asynccnt 0x0" ::: "memory");
        __syncthreads();
        if (kk + 1 < NCHUNK) issue_chunk(kk + 1, buf ^ 1);

        if ((kk & 1) == 0) {
            #pragma unroll
            for (int i = 0; i < 2; ++i)
                #pragma unroll
                for (int j = 0; j < 2; ++j)
                    acc[i][j] = (v8f){};
        }

        v16i a[2], b[2];
        #pragma unroll
        for (int t = 0; t < 2; ++t) {
            int rowA = wm * 32 + 16 * t + laneN;
            #pragma unroll
            for (int h = 0; h < 2; ++h) {
                #pragma unroll
                for (int p = 0; p < 4; ++p) {
                    v2i d = *(const v2i*)&sA[buf][rowA][h * 64 + (p >> 1) * 32 + half * 16 + (p & 1) * 8];
                    a[t][8 * h + 2 * p]     = d[0];
                    a[t][8 * h + 2 * p + 1] = d[1];
                }
            }
            int rowB = wn * 32 + 16 * t + laneN;
            #pragma unroll
            for (int qq = 0; qq < 4; ++qq) {
                v4i d = *(const v4i*)&sB[buf][rowB][qq * 32 + half * 16];
                b[t][4 * qq + 0] = d[0];
                b[t][4 * qq + 1] = d[1];
                b[t][4 * qq + 2] = d[2];
                b[t][4 * qq + 3] = d[3];
            }
        }
        #pragma unroll
        for (int i = 0; i < 2; ++i)
            #pragma unroll
            for (int j = 0; j < 2; ++j)
                acc[i][j] = __builtin_amdgcn_wmma_f32_16x16x128_fp8_fp8(
                    a[i], b[j], (short)0, acc[i][j], false, false);

        if (kk & 1) {
            int g = kk >> 1;
            #pragma unroll
            for (int i = 0; i < 2; ++i) {
                int mrow = mWG + wm * 32 + 16 * i + half * 8;
                float sxr[8];
                #pragma unroll
                for (int r = 0; r < 8; ++r)
                    sxr[r] = sx[(size_t)(mrow + r) * KG + g];
                #pragma unroll
                for (int j = 0; j < 2; ++j) {
                    int ncol = nWG + wn * 32 + 16 * j + laneN;
                    float s = 0.25f * sw[(size_t)ncol * KG + g];
                    #pragma unroll
                    for (int r = 0; r < 8; ++r)
                        facc[i][j][r] += acc[i][j][r] * (sxr[r] * s);
                }
            }
        }
    }

    #pragma unroll
    for (int i = 0; i < 2; ++i) {
        int mrow = mWG + wm * 32 + 16 * i + half * 8;
        #pragma unroll
        for (int j = 0; j < 2; ++j) {
            int ncol = nWG + wn * 32 + 16 * j + laneN;
            float bv = bias[ncol];
            #pragma unroll
            for (int r = 0; r < 8; ++r)
                out[(size_t)(mrow + r) * N + ncol] = facc[i][j][r] + bv;
        }
    }
}

extern "C" void kernel_launch(void* const* d_in, const int* in_sizes, int n_in,
                              void* d_out, int out_size, void* d_ws, size_t ws_size,
                              hipStream_t stream) {
    (void)in_sizes; (void)n_in; (void)out_size; (void)ws_size;
    const float* x    = (const float*)d_in[0];
    const float* w    = (const float*)d_in[1];
    const float* bias = (const float*)d_in[2];
    float* out = (float*)d_out;

    const int M = GEMM_M, N = GEMM_N, K = GEMM_K;

    char* ws = (char*)d_ws;
    unsigned char* qx = (unsigned char*)ws;
    unsigned char* qw = qx + (size_t)M * K;
    float* sx = (float*)(qw + (size_t)N * K);
    float* sw = sx + (size_t)M * KG;

    int gx = (M * K) / 256;
    int gw = (N * K) / 256;
    fp4_quant_kernel<<<gx / 8, 256, 0, stream>>>(x, qx, sx, gx);
    fp4_quant_kernel<<<gw / 8, 256, 0, stream>>>(w, qw, sw, gw);

    dim3 grid(N / 64, M / 128);
    fp4_gemm_wmma_kernel<<<grid, 256, 0, stream>>>(qx, qw, sx, sw, bias, out);
}
